// ScalarDotProductPolicyNetwork_32590211842700
// MI455X (gfx1250) — matrix operand order repaired and checked by reading, not compiled, not run on hardware
//
#include <hip/hip_runtime.h>
#include <hip/hip_bf16.h>

typedef __attribute__((ext_vector_type(16))) _Float16 v16h;
typedef __attribute__((ext_vector_type(8)))  float    v8f;

// ---------- WMMA helpers (gfx1250, wave32) ----------

__device__ __forceinline__ v8f wmma_f16(v16h a, v16h b, v8f c) {
  // D = A(16x32 f16) * B(32x16 f16) + C(16x16 f32)
  return __builtin_amdgcn_wmma_f32_16x16x32_f16(
      /*neg_a=*/false, a, /*neg_b=*/false, b,
      /*c_mod=*/(short)0, c, /*reuse_a=*/false, /*reuse_b=*/false);
}

__device__ __forceinline__ v8f zero8() {
  v8f z = {0.f, 0.f, 0.f, 0.f, 0.f, 0.f, 0.f, 0.f};
  return z;
}

// A fragment (16x32, f16) from LDS row-major matrix [*, ld], tile origin (m0,k0).
// ISA layout: lanes 0-15 row M=lane, halves 0..7 -> K+{0..7}, halves 8..15 -> K+{16..23};
// lanes 16-31 same rows, K offset +8.
__device__ __forceinline__ v16h load_a_lds(const _Float16* M, int ld, int m0, int k0, int lane) {
  const _Float16* p = M + (m0 + (lane & 15)) * ld + k0 + ((lane & 16) ? 8 : 0);
  v16h a;
#pragma unroll
  for (int i = 0; i < 8; ++i) { a[i] = p[i]; a[8 + i] = p[16 + i]; }
  return a;
}

// A fragment from global f32 row-major matrix [*, ld], converted to f16.
__device__ __forceinline__ v16h load_a_gf32(const float* M, int ld, int m0, int k0, int lane) {
  const float* p = M + (m0 + (lane & 15)) * ld + k0 + ((lane & 16) ? 8 : 0);
  v16h a;
#pragma unroll
  for (int i = 0; i < 8; ++i) { a[i] = (_Float16)p[i]; a[8 + i] = (_Float16)p[16 + i]; }
  return a;
}

// B fragment (32x16): B[k][n] = W[n][k], W row-major f16 GLOBAL [*, ldW].
// ISA layout: lanes 0-15 col N=lane, halves 0..7 -> K+{0..7}, halves 8..15 -> K+{16..23};
// lanes 16-31 same cols, K offset +8.
__device__ __forceinline__ v16h load_b_wT_gh(const _Float16* W, int ldW, int k0, int n0, int lane) {
  const _Float16* p = W + (n0 + (lane & 15)) * ldW + k0 + ((lane & 16) ? 8 : 0);
  v16h b;
#pragma unroll
  for (int i = 0; i < 8; ++i) { b[i] = p[i]; b[8 + i] = p[16 + i]; }
  return b;
}

// Same but from an LDS f16 matrix Mt (B[k][n] = Mt[n][k]): same K layout as above.
__device__ __forceinline__ v16h load_b_wT_lds(const _Float16* Mt, int ld, int k0, int n0, int lane) {
  const _Float16* p = Mt + (n0 + (lane & 15)) * ld + k0 + ((lane & 16) ? 8 : 0);
  v16h b;
#pragma unroll
  for (int i = 0; i < 8; ++i) { b[i] = p[i]; b[8 + i] = p[16 + i]; }
  return b;
}

// E(LDS f16 64x64) @ W^T (W global f16 [64,64]) -> 4 col tiles of 16x16, K=64.
__device__ __forceinline__ void gemm_EW(const _Float16* sA, const _Float16* W,
                                        v8f acc[4], int m0, int lane) {
#pragma unroll
  for (int c = 0; c < 4; ++c) acc[c] = zero8();
#pragma unroll
  for (int kb = 0; kb < 2; ++kb) {
    v16h a = load_a_lds(sA, 64, m0, kb * 32, lane);
#pragma unroll
    for (int c = 0; c < 4; ++c) {
      v16h b = load_b_wT_gh(W, 64, kb * 32, c * 16, lane);
      acc[c] = wmma_f16(a, b, acc[c]);
    }
  }
}

// ---------- One-shot weight f32->f16 conversion into workspace ----------
// ws layout (halves): [0,8192) W_emb  [8192,12288) W_k  [12288,16384) W_q
//                     [16384,20480) W_v  [20480,24576) W_f1  [24576,25600) W_f2
#define WS_EMB 0
#define WS_K   8192
#define WS_Q   12288
#define WS_V   16384
#define WS_F1  20480
#define WS_F2  24576
#define WS_TOT 25600

__global__ __launch_bounds__(256)
void convert_weights_kernel(const float* __restrict__ W_emb, const float* __restrict__ W_k,
                            const float* __restrict__ W_q,   const float* __restrict__ W_v,
                            const float* __restrict__ W_f1,  const float* __restrict__ W_f2,
                            _Float16* __restrict__ ws) {
  int i = blockIdx.x * blockDim.x + threadIdx.x;
  if (i >= WS_TOT) return;
  float v;
  if      (i < WS_K)  v = W_emb[i - WS_EMB];
  else if (i < WS_Q)  v = W_k  [i - WS_K];
  else if (i < WS_V)  v = W_q  [i - WS_Q];
  else if (i < WS_F1) v = W_v  [i - WS_V];
  else if (i < WS_F2) v = W_f1 [i - WS_F1];
  else                v = W_f2 [i - WS_F2];
  ws[i] = (_Float16)v;
}

// ---------- Main kernel: one block per batch element, 4 waves (wave32) ----------

__global__ __launch_bounds__(128)
void policy_net_kernel(const float* __restrict__ states,     // [B,64,128]
                       const _Float16* __restrict__ ws,      // f16 weights (see layout)
                       const float* __restrict__ b_emb,      // [64]
                       float* __restrict__ out_policy,       // [B,64,16]
                       float* __restrict__ out_weight)       // [B,64,64]
{
  const _Float16* hW_emb = ws + WS_EMB;  // [64][128]
  const _Float16* hW_k   = ws + WS_K;    // [64][64]
  const _Float16* hW_q   = ws + WS_Q;    // [64][64]
  const _Float16* hW_v   = ws + WS_V;    // [64][64]
  const _Float16* hW_f1  = ws + WS_F1;   // [64][64]
  const _Float16* hW_f2  = ws + WS_F2;   // [16][64]

  // 48KB LDS arena with phase-aliased buffers.
  __shared__ __align__(16) unsigned char smem[49152];
  _Float16* sE  = (_Float16*)(smem);           // [64][64] f16
  _Float16* sK  = (_Float16*)(smem +  8192);   // [64][64] f16
  _Float16* sQ  = (_Float16*)(smem + 16384);   // [64][64] f16
  _Float16* sVT = (_Float16*)(smem + 24576);   // [d][j]   f16 (V transposed)
  float*    sSc = (float*)   (smem + 32768);   // [64][64] f32 scores
  _Float16* sWf = (_Float16*)(smem);           // alias sE : softmax weights f16
  _Float16* sNd = (_Float16*)(smem +  8192);   // alias sK : node (unscaled)
  _Float16* sH  = (_Float16*)(smem + 16384);   // alias sQ : hidden
  float*    sLg = (float*)   (smem + 32768);   // alias sSc: [64][16] logits

  const int b    = blockIdx.x;
  const int tid  = threadIdx.x;
  const int wave = tid >> 5;
  const int lane = tid & 31;
  const int m0   = wave * 16;                  // wave's 16-row stripe of N=64
  const int nD   = lane & 15;                  // D-tile column for this lane
  const int mD   = m0 + ((lane & 16) ? 8 : 0); // D-tile first row for this lane

  const float* S = states + (size_t)b * 64 * 128;

  // ---- 1) E = relu(S @ W_emb^T + b_emb)   (64x64, K=128)
  {
    v8f acc[4];
#pragma unroll
    for (int c = 0; c < 4; ++c) acc[c] = zero8();
#pragma unroll
    for (int kb = 0; kb < 4; ++kb) {
      v16h a = load_a_gf32(S, 128, m0, kb * 32, lane);   // states read once from HBM
#pragma unroll
      for (int c = 0; c < 4; ++c) {
        v16h bf = load_b_wT_gh(hW_emb, 128, kb * 32, c * 16, lane);
        acc[c] = wmma_f16(a, bf, acc[c]);
      }
    }
#pragma unroll
    for (int c = 0; c < 4; ++c) {
      int n = c * 16 + nD;
      float bias = b_emb[n];
#pragma unroll
      for (int r = 0; r < 8; ++r) {
        float e = acc[c][r] + bias;
        sE[(mD + r) * 64 + n] = (_Float16)(e > 0.f ? e : 0.f);
      }
    }
  }
  __syncthreads();

  // ---- 2) K = E@Wk^T, Q = E@Wq^T, V = tanh(E@Wv^T) (stored transposed)
  {
    v8f t[4];
    gemm_EW(sE, hW_k, t, m0, lane);
#pragma unroll
    for (int c = 0; c < 4; ++c)
#pragma unroll
      for (int r = 0; r < 8; ++r)
        sK[(mD + r) * 64 + c * 16 + nD] = (_Float16)t[c][r];

    gemm_EW(sE, hW_q, t, m0, lane);
#pragma unroll
    for (int c = 0; c < 4; ++c)
#pragma unroll
      for (int r = 0; r < 8; ++r)
        sQ[(mD + r) * 64 + c * 16 + nD] = (_Float16)t[c][r];

    gemm_EW(sE, hW_v, t, m0, lane);
#pragma unroll
    for (int c = 0; c < 4; ++c)
#pragma unroll
      for (int r = 0; r < 8; ++r)
        sVT[(c * 16 + nD) * 64 + (mD + r)] = (_Float16)tanhf(t[c][r]);
  }
  __syncthreads();

  // ---- 3) scores[i][j] = (K[i,:] . Q[j,:]) / sqrt(64)
  {
    v8f t[4];
#pragma unroll
    for (int c = 0; c < 4; ++c) t[c] = zero8();
#pragma unroll
    for (int kb = 0; kb < 2; ++kb) {
      v16h a = load_a_lds(sK, 64, m0, kb * 32, lane);
#pragma unroll
      for (int c = 0; c < 4; ++c) {
        v16h bf = load_b_wT_lds(sQ, 64, kb * 32, c * 16, lane);  // B[d][j]=Q[j][d]
        t[c] = wmma_f16(a, bf, t[c]);
      }
    }
#pragma unroll
    for (int c = 0; c < 4; ++c)
#pragma unroll
      for (int r = 0; r < 8; ++r)
        sSc[(mD + r) * 64 + c * 16 + nD] = t[c][r] * 0.125f;  // 1/sqrt(D)
  }
  __syncthreads();

  // ---- 4) weight = softmax(scores, axis=j); emit f32 to HBM + f16 to LDS
  if (tid < 64) {
    float* row = sSc + tid * 64;
    float mx = row[0];
    for (int j = 1; j < 64; ++j) mx = fmaxf(mx, row[j]);
    float sum = 0.f;
    for (int j = 0; j < 64; ++j) { float ev = __expf(row[j] - mx); row[j] = ev; sum += ev; }
    float inv = 1.f / sum;
    float* wout = out_weight + ((size_t)b * 64 + tid) * 64;
    for (int j = 0; j < 64; ++j) {
      float w = row[j] * inv;
      wout[j] = w;
      sWf[tid * 64 + j] = (_Float16)w;
    }
  }
  __syncthreads();

  // ---- 5) node' = weight @ V   (1/N scaling deferred to step 6 epilogue)
  {
    v8f t[4];
#pragma unroll
    for (int c = 0; c < 4; ++c) t[c] = zero8();
#pragma unroll
    for (int kb = 0; kb < 2; ++kb) {
      v16h a = load_a_lds(sWf, 64, m0, kb * 32, lane);
#pragma unroll
      for (int c = 0; c < 4; ++c) {
        v16h bf = load_b_wT_lds(sVT, 64, kb * 32, c * 16, lane);  // B[j][d]=V[j][d]
        t[c] = wmma_f16(a, bf, t[c]);
      }
    }
#pragma unroll
    for (int c = 0; c < 4; ++c)
#pragma unroll
      for (int r = 0; r < 8; ++r)
        sNd[(mD + r) * 64 + c * 16 + nD] = (_Float16)t[c][r];
  }
  __syncthreads();

  // ---- 6) h = leaky_relu( (node'/N) @ W_f1^T ), slope 0.01
  {
    v8f t[4];
#pragma unroll
    for (int c = 0; c < 4; ++c) t[c] = zero8();
#pragma unroll
    for (int kb = 0; kb < 2; ++kb) {
      v16h a = load_a_lds(sNd, 64, m0, kb * 32, lane);
#pragma unroll
      for (int c = 0; c < 4; ++c) {
        v16h bf = load_b_wT_gh(hW_f1, 64, kb * 32, c * 16, lane);
        t[c] = wmma_f16(a, bf, t[c]);
      }
    }
#pragma unroll
    for (int c = 0; c < 4; ++c)
#pragma unroll
      for (int r = 0; r < 8; ++r) {
        float x = t[c][r] * (1.f / 64.f);   // fold mean-over-j here (linear)
        sH[(mD + r) * 64 + c * 16 + nD] = (_Float16)(x > 0.f ? x : 0.01f * x);
      }
  }
  __syncthreads();

  // ---- 7) logits = h @ W_f2^T   (64x16, K=64 -> single col tile)
  {
    v8f t = zero8();
#pragma unroll
    for (int kb = 0; kb < 2; ++kb) {
      v16h a  = load_a_lds(sH, 64, m0, kb * 32, lane);
      v16h bf = load_b_wT_gh(hW_f2, 64, kb * 32, 0, lane);
      t = wmma_f16(a, bf, t);
    }
#pragma unroll
    for (int r = 0; r < 8; ++r)
      sLg[(mD + r) * 16 + nD] = t[r];
  }
  __syncthreads();

  // ---- 8) policy = softmax(logits, axis=a)
  if (tid < 64) {
    const float* row = sLg + tid * 16;
    float mx = row[0];
    for (int j = 1; j < 16; ++j) mx = fmaxf(mx, row[j]);
    float ex[16];
    float sum = 0.f;
    for (int j = 0; j < 16; ++j) { ex[j] = __expf(row[j] - mx); sum += ex[j]; }
    float inv = 1.f / sum;
    float* pout = out_policy + ((size_t)b * 64 + tid) * 16;
    for (int j = 0; j < 16; ++j) pout[j] = ex[j] * inv;
  }
}

extern "C" void kernel_launch(void* const* d_in, const int* in_sizes, int n_in,
                              void* d_out, int out_size, void* d_ws, size_t ws_size,
                              hipStream_t stream) {
  const float* states = (const float*)d_in[0];
  const float* W_emb  = (const float*)d_in[1];
  const float* b_emb  = (const float*)d_in[2];
  const float* W_k    = (const float*)d_in[3];
  const float* W_q    = (const float*)d_in[4];
  const float* W_v    = (const float*)d_in[5];
  const float* W_f1   = (const float*)d_in[6];
  const float* W_f2   = (const float*)d_in[7];

  const int B = in_sizes[0] / (64 * 128);     // 4096

  float* out        = (float*)d_out;
  float* out_policy = out;                              // [B,64,16] first
  float* out_weight = out + (size_t)B * 64 * 16;        // [B,64,64] second

  _Float16* ws = (_Float16*)d_ws;                       // 25600 halves = 50KB

  // One-shot f32->f16 weight conversion (reruns every call; deterministic).
  convert_weights_kernel<<<dim3((WS_TOT + 255) / 256), dim3(256), 0, stream>>>(
      W_emb, W_k, W_q, W_v, W_f1, W_f2, ws);

  policy_net_kernel<<<dim3(B), dim3(128), 0, stream>>>(
      states, ws, b_emb, out_policy, out_weight);
}
